// PermutationMatrix_17918603559094
// MI455X (gfx1250) — matrix operand order repaired and checked by reading, not compiled, not run on hardware
//
#include <hip/hip_runtime.h>

#define GN   4096
#define TEMP 100.0f
#define ITERS 20
#define SA   64.0f
#define SB   4096.0f
#define CSCALE (1.0f / (SA * SB))

// GEMM tiling: block tile 128(M) x 256(N), K-step 32, 8 wave32 as 2x4, wave tile 64x64
#define BM 128
#define BN 256
#define BK 32
#define LDK 40   // padded LDS row (40 halfs = 80B = 20 banks -> conflict-free frags)

typedef __attribute__((ext_vector_type(2)))  _Float16 v2h;
typedef __attribute__((ext_vector_type(8)))  _Float16 v8h;
typedef __attribute__((ext_vector_type(16))) _Float16 v16h;
typedef __attribute__((ext_vector_type(8)))  float    v8f;

// ---------------- global max (two stage) ----------------
__global__ __launch_bounds__(256) void max_partial_k(const float* __restrict__ X,
                                                     float* __restrict__ bmax) {
  __shared__ float s[256];
  float m = -3.402823466e38f;
  const size_t total = (size_t)GN * GN;
  for (size_t idx = (size_t)blockIdx.x * 256 + threadIdx.x; idx < total;
       idx += (size_t)gridDim.x * 256)
    m = fmaxf(m, X[idx]);
  s[threadIdx.x] = m;
  __syncthreads();
  for (int off = 128; off > 0; off >>= 1) {
    if ((int)threadIdx.x < off) s[threadIdx.x] = fmaxf(s[threadIdx.x], s[threadIdx.x + off]);
    __syncthreads();
  }
  if (threadIdx.x == 0) bmax[blockIdx.x] = s[0];
}

__global__ __launch_bounds__(256) void max_final_k(const float* __restrict__ bmax,
                                                   float* __restrict__ maxv, int n) {
  __shared__ float s[256];
  float m = -3.402823466e38f;
  for (int i = threadIdx.x; i < n; i += 256) m = fmaxf(m, bmax[i]);
  s[threadIdx.x] = m;
  __syncthreads();
  for (int off = 128; off > 0; off >>= 1) {
    if ((int)threadIdx.x < off) s[threadIdx.x] = fmaxf(s[threadIdx.x], s[threadIdx.x + off]);
    __syncthreads();
  }
  if (threadIdx.x == 0) maxv[0] = s[0];
}

// ---------------- Mh = (f16) exp(T*(x - max)) ----------------
// values in [exp(-3.125), 1] ~ [0.044, 1]: comfortably in f16 normal range
__global__ __launch_bounds__(256) void exp_k(const float* __restrict__ X,
                                             const float* __restrict__ maxv,
                                             _Float16* __restrict__ Mh) {
  const float mx = maxv[0];
  const int total8 = (GN / 8) * GN;
  const float4* __restrict__ X4 = (const float4*)X;
  v8h* __restrict__ M8 = (v8h*)Mh;
  for (int idx = blockIdx.x * 256 + threadIdx.x; idx < total8; idx += gridDim.x * 256) {
    const float4 x0 = X4[2 * idx];
    const float4 x1 = X4[2 * idx + 1];
    v8h o;
    o[0] = (_Float16)expf(TEMP * (x0.x - mx));
    o[1] = (_Float16)expf(TEMP * (x0.y - mx));
    o[2] = (_Float16)expf(TEMP * (x0.z - mx));
    o[3] = (_Float16)expf(TEMP * (x0.w - mx));
    o[4] = (_Float16)expf(TEMP * (x1.x - mx));
    o[5] = (_Float16)expf(TEMP * (x1.y - mx));
    o[6] = (_Float16)expf(TEMP * (x1.z - mx));
    o[7] = (_Float16)expf(TEMP * (x1.w - mx));
    M8[idx] = o;
  }
}

// ---------------- Sinkhorn on scaling vectors only ----------------
__global__ __launch_bounds__(256) void init_uv_k(float* __restrict__ u, float* __restrict__ v) {
  int i = blockIdx.x * 256 + threadIdx.x;
  if (i < GN) { u[i] = 1.0f; v[i] = 1.0f; }
}

// u_i = 1 / sum_j Mh[i][j] * v[j] ; also zero colsum for the column phase
__global__ __launch_bounds__(256) void sk_row_k(const _Float16* __restrict__ Mh,
                                                const float* __restrict__ v,
                                                float* __restrict__ u,
                                                float* __restrict__ colsum) {
  __shared__ float s[256];
  const int i = blockIdx.x;
  const v8h* __restrict__ row = (const v8h*)(Mh + (size_t)i * GN);
  float acc = 0.0f;
  for (int c = threadIdx.x; c < GN / 8; c += 256) {
    const v8h m = row[c];
    const float* vv = v + c * 8;
#pragma unroll
    for (int e = 0; e < 8; ++e) acc += (float)m[e] * vv[e];
  }
  s[threadIdx.x] = acc;
  __syncthreads();
  for (int off = 128; off > 0; off >>= 1) {
    if ((int)threadIdx.x < off) s[threadIdx.x] += s[threadIdx.x + off];
    __syncthreads();
  }
  if (threadIdx.x == 0) { u[i] = 1.0f / s[0]; colsum[i] = 0.0f; }
}

// colsum[j] += sum over a 32-row chunk of Mh[r][j] * u[r]  (2 columns / thread)
__global__ __launch_bounds__(256) void sk_col_partial_k(const _Float16* __restrict__ Mh,
                                                        const float* __restrict__ u,
                                                        float* __restrict__ colsum) {
  const int j2 = (blockIdx.x * 256 + threadIdx.x) * 2;
  const int r0 = blockIdx.y * 32;
  float a0 = 0.0f, a1 = 0.0f;
  for (int r = r0; r < r0 + 32; ++r) {
    const v2h m = *(const v2h*)(Mh + (size_t)r * GN + j2);
    const float ur = u[r];
    a0 += (float)m.x * ur;
    a1 += (float)m.y * ur;
  }
  atomicAdd(&colsum[j2], a0);
  atomicAdd(&colsum[j2 + 1], a1);
}

__global__ __launch_bounds__(256) void sk_col_inv_k(const float* __restrict__ colsum,
                                                    float* __restrict__ v) {
  const int j = blockIdx.x * 256 + threadIdx.x;
  v[j] = 1.0f / colsum[j];
}

// ---------------- GEMM operand builders ----------------
// B16[t][k] = SB * u_t * Mh[t][k] * v_k   (f16)
__global__ __launch_bounds__(256) void build_B_k(const _Float16* __restrict__ Mh,
                                                 const float* __restrict__ u,
                                                 const float* __restrict__ v,
                                                 _Float16* __restrict__ B16) {
  const int k = blockIdx.x * 256 + threadIdx.x;
  const int t = blockIdx.y;
  const float val = SB * u[t] * (float)Mh[(size_t)t * GN + k] * v[k];
  B16[(size_t)t * GN + k] = (_Float16)val;
}

// A16[i][k] = SA * u_i * prefixsum_{j<=k}( Mh[i][j] * v[j] )   (f16)
// NOTE: A16 may alias Mh — each element is read and later written by the SAME thread.
__global__ __launch_bounds__(256) void build_A_k(const _Float16* __restrict__ Mh,
                                                 const float* __restrict__ u,
                                                 const float* __restrict__ v,
                                                 _Float16* __restrict__ A16) {
  __shared__ float s[256];
  __shared__ float carry;
  const int i = blockIdx.x;
  const int tid = threadIdx.x;
  if (tid == 0) carry = 0.0f;
  __syncthreads();
  const float ui = SA * u[i];
  for (int c = 0; c < GN / 256; ++c) {
    const int j = c * 256 + tid;
    s[tid] = (float)Mh[(size_t)i * GN + j] * v[j];
    __syncthreads();
    // inclusive Hillis-Steele scan over 256 entries
    for (int off = 1; off < 256; off <<= 1) {
      float t = (tid >= off) ? s[tid - off] : 0.0f;
      __syncthreads();
      s[tid] += t;
      __syncthreads();
    }
    const float inc = s[tid];
    const float cl = carry;
    A16[(size_t)i * GN + j] = (_Float16)(ui * (inc + cl));
    __syncthreads();
    if (tid == 255) carry = cl + inc;
    __syncthreads();
  }
}

// ---------------- C = (A * B^T) * CSCALE via v_wmma_f32_16x16x32_f16 ----------------
// 256 threads = 8 wave32 arranged 2(M) x 4(N); wave tile 64x64 = 4x4 WMMA tiles.
// 16 ds_load_b128 per 16 WMMAs per K-step (1.0 LDS loads / WMMA).
__global__ __launch_bounds__(256) void gemm_ABt_k(const _Float16* __restrict__ A,
                                                  const _Float16* __restrict__ B,
                                                  float* __restrict__ C) {
  __shared__ _Float16 As[BM][LDK];
  __shared__ _Float16 Bs[BN][LDK];

  const int tid  = threadIdx.x;
  const int lane = tid & 31;
  const int w    = tid >> 5;   // 0..7
  const int wm   = w & 1;      // 2 waves along M (64 rows each)
  const int wn   = w >> 1;     // 4 waves along N (64 cols each)

  const int rowBase = blockIdx.y * BM;
  const int colBase = blockIdx.x * BN;

  // tile-load assignment: A: 2 threads/row (16 halfs each); B: 1 thread/row (32 halfs)
  const int alr = tid >> 1;          // 0..127
  const int alc = (tid & 1) * 16;    // 0 or 16

  v8f acc[4][4];
  for (int a = 0; a < 4; ++a)
    for (int b = 0; b < 4; ++b) acc[a][b] = {};

  const int l16 = lane & 15;
  const int h   = lane >> 4;

  for (int k0 = 0; k0 < GN; k0 += BK) {
    const _Float16* ga = A + (size_t)(rowBase + alr) * GN + k0 + alc;
    const _Float16* gb = B + (size_t)(colBase + tid) * GN + k0;
    float4 a0 = *(const float4*)ga;
    float4 a1 = *(const float4*)(ga + 8);
    float4 b0 = *(const float4*)gb;
    float4 b1 = *(const float4*)(gb + 8);
    float4 b2 = *(const float4*)(gb + 16);
    float4 b3 = *(const float4*)(gb + 24);
    __syncthreads();
    *(float4*)&As[alr][alc]     = a0;
    *(float4*)&As[alr][alc + 8] = a1;
    *(float4*)&Bs[tid][0]  = b0;
    *(float4*)&Bs[tid][8]  = b1;
    *(float4*)&Bs[tid][16] = b2;
    *(float4*)&Bs[tid][24] = b3;
    __syncthreads();

    // B fragments (lane half h holds K = 8h..8h+7 then 16+8h..16+8h+7, matching A)
    v16h bfrag[4];
#pragma unroll
    for (int tn = 0; tn < 4; ++tn) {
      const _Float16* bp = &Bs[wn * 64 + tn * 16 + l16][0];
      v8h lo = *(const v8h*)(bp + 8 * h);
      v8h hi = *(const v8h*)(bp + 16 + 8 * h);
      bfrag[tn] = __builtin_shufflevector(lo, hi, 0, 1, 2, 3, 4, 5, 6, 7,
                                          8, 9, 10, 11, 12, 13, 14, 15);
    }
#pragma unroll
    for (int tm = 0; tm < 4; ++tm) {
      // A fragment (ISA: lane half h holds K = 8h..8h+7 then 16+8h..16+8h+7)
      const _Float16* ap = &As[wm * 64 + tm * 16 + l16][0];
      v8h lo = *(const v8h*)(ap + 8 * h);
      v8h hi = *(const v8h*)(ap + 16 + 8 * h);
      v16h afrag = __builtin_shufflevector(lo, hi, 0, 1, 2, 3, 4, 5, 6, 7,
                                           8, 9, 10, 11, 12, 13, 14, 15);
#pragma unroll
      for (int tn = 0; tn < 4; ++tn)
        acc[tm][tn] = __builtin_amdgcn_wmma_f32_16x16x32_f16(
            false, afrag, false, bfrag[tn], (short)0, acc[tm][tn], false, false);
    }
  }

  // epilogue: D layout — VGPR g: M = 8*(lane/16) + g, N = lane%16
#pragma unroll
  for (int tm = 0; tm < 4; ++tm) {
#pragma unroll
    for (int tn = 0; tn < 4; ++tn) {
      const int col = colBase + wn * 64 + tn * 16 + l16;
#pragma unroll
      for (int g = 0; g < 8; ++g) {
        const int row = rowBase + wm * 64 + tm * 16 + h * 8 + g;
        C[(size_t)row * GN + col] = acc[tm][tn][g] * CSCALE;
      }
    }
  }
}

extern "C" void kernel_launch(void* const* d_in, const int* in_sizes, int n_in,
                              void* d_out, int out_size, void* d_ws, size_t ws_size,
                              hipStream_t stream) {
  (void)in_sizes; (void)n_in; (void)out_size; (void)ws_size;
  const float* X = (const float*)d_in[0];
  float* C = (float*)d_out;

  char* w = (char*)d_ws;
  const size_t SZH = (size_t)GN * GN * sizeof(_Float16);  // 32 MB
  _Float16* Mh    = (_Float16*)w;        // f16 exp matrix; later overwritten in place by A16
  _Float16* A16   = Mh;                  // alias (see build_A_k)
  _Float16* B16   = (_Float16*)(w + SZH);
  float* u        = (float*)(w + 2 * SZH);
  float* v        = u + GN;
  float* colsum   = v + GN;
  float* bmax     = colsum + GN;
  float* maxv     = bmax + 1024;

  // 1) global max, then Mh = (f16) exp(T*(x-max))
  max_partial_k<<<1024, 256, 0, stream>>>(X, bmax);
  max_final_k<<<1, 256, 0, stream>>>(bmax, maxv, 1024);
  exp_k<<<2048, 256, 0, stream>>>(X, maxv, Mh);

  // 2) Sinkhorn iterations on scaling vectors u, v only (2 f16 matvecs / iter)
  init_uv_k<<<GN / 256, 256, 0, stream>>>(u, v);
  for (int it = 0; it < ITERS; ++it) {
    sk_row_k<<<GN, 256, 0, stream>>>(Mh, v, u, colsum);
    sk_col_partial_k<<<dim3(GN / 512, GN / 32), 256, 0, stream>>>(Mh, u, colsum);
    sk_col_inv_k<<<GN / 256, 256, 0, stream>>>(colsum, v);
  }

  // 3) out = (m L m^T)^T = (m L^T) m^T = A * B^T; A = scaled row prefix sums, B = scaled m
  build_B_k<<<dim3(GN / 256, GN), 256, 0, stream>>>(Mh, u, v, B16);   // must run before A overwrites Mh
  build_A_k<<<GN, 256, 0, stream>>>(Mh, u, v, A16);                   // in-place over Mh
  gemm_ABt_k<<<dim3(GN / BN, GN / BM), 256, 0, stream>>>(A16, B16, C);
}
